// Net_26620207300991
// MI455X (gfx1250) — compile-verified, matrix operand order checked by reading, not run on hardware
//
#include <hip/hip_runtime.h>
#include <hip/hip_bf16.h>
#include <math.h>

typedef __attribute__((ext_vector_type(16))) __bf16 v16bf;
typedef __attribute__((ext_vector_type(8)))  float  v8f;

// ---------------------------------------------------------------------------
// Problem constants
// ---------------------------------------------------------------------------
#define BG   64
#define NPG  1024
#define EDG  (64 * 1024 * 16)   // 1,048,576 edges
#define N1   922                // ceil(0.9*1024)
#define N2   461                // ceil(0.5*922)
#define N3   231                // ceil(0.5*461)

// ---------------------------------------------------------------------------
// Weight pre-pack into WMMA B-fragment order (bf16, zero-padded):
//   Wp[((colTile*nCH + ch)*32 + lane)*16 + e]
// with (k,n) per CDNA5 16-bit B striping: n = colTile*16 + lane%16,
//   i=e/2: kbase = (i<4 ? 2i : 16+2(i-4)); k = ch*32 + kbase + 8*(lane/16) + e%2
// ---------------------------------------------------------------------------
__global__ void pack_w_k(const float* __restrict__ W, __bf16* __restrict__ Wp,
                         int K, int N, int KP, int NP) {
  const int idx = blockIdx.x * blockDim.x + threadIdx.x;
  const int nCH = KP >> 5;
  const int total = (NP >> 4) * nCH * 512;
  if (idx >= total) return;
  const int e    = idx & 15;
  const int lane = (idx >> 4) & 31;
  const int rest = idx >> 9;
  const int ch   = rest % nCH;
  const int ct   = rest / nCH;
  const int i    = e >> 1;
  const int kbase = (i < 4) ? (2 * i) : (16 + 2 * (i - 4));
  const int k = ch * 32 + kbase + (lane >> 4) * 8 + (e & 1);
  const int n = ct * 16 + (lane & 15);
  const int kc = (k < K) ? k : (K - 1);
  const int nc = (n < N) ? n : (N - 1);
  float v = W[(size_t)kc * N + nc];          // always in-range (clamped)
  v = (k < K && n < N) ? v : 0.0f;           // select, no exec branch
  Wp[idx] = (__bf16)v;
}

__global__ void pad_bias_k(const float* __restrict__ b, float* __restrict__ bp,
                           int N, int NP) {
  const int n = threadIdx.x;
  if (n >= NP) return;
  float v = b[(n < N) ? n : (N - 1)];
  bp[n] = (n < N) ? v : 0.0f;
}

__global__ void pad_x_k(const float* __restrict__ x, float* __restrict__ xp,
                        int total) {  // C=31 -> CP=32
  const int i = blockIdx.x * blockDim.x + threadIdx.x;
  if (i >= total) return;
  const int c = i & 31;
  const int r = i >> 5;
  float v = x[(size_t)r * 31 + ((c < 31) ? c : 30)];
  xp[i] = (c < 31) ? v : 0.0f;
}

// ---------------------------------------------------------------------------
// Guard-free WMMA GEMM on padded operands:
//   Y[M,NP] = A[M,KP] @ W[KP,NP] + biasP[NP]
// grid = (NP/16, M/64), block = 128 (4 waves, one 16x16 tile each).
// A: 4x global_load_b128 per 32-K chunk; B: one 32B fragment load per chunk.
// ---------------------------------------------------------------------------
__global__ void gemm_wmma_bf16(const float* __restrict__ A,
                               const __bf16* __restrict__ Wp,
                               const float* __restrict__ biasP,
                               float* __restrict__ Y,
                               int KP, int NP) {
  const int lane = threadIdx.x & 31;
  const int wave = threadIdx.x >> 5;
  const int col0 = blockIdx.x * 16;
  const int row0 = (blockIdx.y * 4 + wave) * 16;
  const int half = lane >> 4;
  const int l15  = lane & 15;
  const int nCH  = KP >> 5;

  const float*  arow = A + (size_t)(row0 + l15) * KP + half * 8;
  const __bf16* wp   = Wp + (size_t)blockIdx.x * nCH * 512 + lane * 16;

  v8f acc = {};
  for (int ch = 0; ch < nCH; ++ch) {
    const float* ap = arow + ch * 32;
    float4 a0 = *(const float4*)(ap);
    float4 a1 = *(const float4*)(ap + 4);
    float4 a2 = *(const float4*)(ap + 16);
    float4 a3 = *(const float4*)(ap + 20);
    v16bf av;
    av[0]  = (__bf16)a0.x; av[1]  = (__bf16)a0.y; av[2]  = (__bf16)a0.z; av[3]  = (__bf16)a0.w;
    av[4]  = (__bf16)a1.x; av[5]  = (__bf16)a1.y; av[6]  = (__bf16)a1.z; av[7]  = (__bf16)a1.w;
    av[8]  = (__bf16)a2.x; av[9]  = (__bf16)a2.y; av[10] = (__bf16)a2.z; av[11] = (__bf16)a2.w;
    av[12] = (__bf16)a3.x; av[13] = (__bf16)a3.y; av[14] = (__bf16)a3.z; av[15] = (__bf16)a3.w;
    v16bf bv = *(const v16bf*)(wp + (size_t)ch * 512);
    acc = __builtin_amdgcn_wmma_f32_16x16x32_bf16(
        false, av, false, bv, (short)0, acc, false, false);
  }

  const int ncol = col0 + l15;
  const float bb = biasP[ncol];
#pragma unroll
  for (int r = 0; r < 8; ++r) {
    Y[(size_t)(row0 + r + half * 8) * NP + ncol] = acc[r] + bb;
  }
}

// ---------------------------------------------------------------------------
// Edges
// ---------------------------------------------------------------------------
__global__ void init_edges_k(const int* __restrict__ ei, int* src, int* dst,
                             int* mask, int E) {
  int e = blockIdx.x * blockDim.x + threadIdx.x;
  if (e >= E) return;
  src[e] = ei[e];
  dst[e] = ei[E + e];
  mask[e] = 1;
}

__global__ void edge_aggregate_k(const float* __restrict__ Kp,
                                 const float* __restrict__ Qp,
                                 const float* __restrict__ Vp,
                                 const int* __restrict__ src,
                                 const int* __restrict__ dst,
                                 const int* __restrict__ mask,
                                 float* __restrict__ out,
                                 int E, int C, int CP) {
  int e = blockIdx.x * blockDim.x + threadIdx.x;
  if (e >= E) return;
  if (!mask[e]) return;
  const int s = src[e], d = dst[e];
  const float* kd = Kp + (size_t)d * CP;
  const float* qs = Qp + (size_t)s * CP;
  const float* vs = Vp + (size_t)s * CP;
  float* od = out + (size_t)d * CP;
  for (int c = 0; c < C; ++c) {
    float g = 1.0f / (1.0f + expf(-(kd[c] + qs[c])));
    atomicAdd(&od[c], g * vs[c]);
  }
}

__global__ void edge_remap_k(int* src, int* dst, int* mask,
                             const int* __restrict__ newidx, int E) {
  int e = blockIdx.x * blockDim.x + threadIdx.x;
  if (e >= E) return;
  int ns = newidx[src[e]];
  int nd = newidx[dst[e]];
  int valid = mask[e] && (ns >= 0) && (nd >= 0);
  src[e] = valid ? ns : 0;
  dst[e] = valid ? nd : 0;
  mask[e] = valid;
}

// ---------------------------------------------------------------------------
// BatchNorm (train-mode, biased var) + ReLU; padded cols forced to 0
// ---------------------------------------------------------------------------
__global__ void bn_stats_k(const float* __restrict__ h, int M, int CP,
                           float* mu, float* rstd) {
  __shared__ float rs[256], rq[256];
  const int c = blockIdx.x;
  float s = 0.0f, q = 0.0f;
  for (int i = threadIdx.x; i < M; i += blockDim.x) {
    float v = h[(size_t)i * CP + c];
    s += v; q += v * v;
  }
  rs[threadIdx.x] = s; rq[threadIdx.x] = q;
  __syncthreads();
  for (int st = 128; st > 0; st >>= 1) {
    if ((int)threadIdx.x < st) {
      rs[threadIdx.x] += rs[threadIdx.x + st];
      rq[threadIdx.x] += rq[threadIdx.x + st];
    }
    __syncthreads();
  }
  if (threadIdx.x == 0) {
    float m = rs[0] / (float)M;
    float var = rq[0] / (float)M - m * m;
    mu[c] = m;
    rstd[c] = rsqrtf(var + 1e-5f);
  }
}

__global__ void bn_apply_relu_k(float* __restrict__ h,
                                const float* __restrict__ mu,
                                const float* __restrict__ rstd,
                                const float* __restrict__ g,
                                const float* __restrict__ beta,
                                int total, int C, int CP) {
  int i = blockIdx.x * blockDim.x + threadIdx.x;
  if (i >= total) return;
  int c = i & (CP - 1);                 // CP is a power of two
  int cc = (c < C) ? c : (C - 1);       // clamp param reads (in-range)
  float v = (h[i] - mu[cc]) * rstd[cc] * g[cc] + beta[cc];
  v = fmaxf(v, 0.0f);
  h[i] = (c < C) ? v : 0.0f;            // keep padding at exactly 0
}

// ---------------------------------------------------------------------------
// TopK pooling
// ---------------------------------------------------------------------------
__global__ void vec_norm_k(const float* __restrict__ w, int C, float* out) {
  __shared__ float rs[256];
  float s = 0.0f;
  for (int i = threadIdx.x; i < C; i += 256) s += w[i] * w[i];
  rs[threadIdx.x] = s;
  __syncthreads();
  for (int st = 128; st > 0; st >>= 1) {
    if ((int)threadIdx.x < st) rs[threadIdx.x] += rs[threadIdx.x + st];
    __syncthreads();
  }
  if (threadIdx.x == 0) out[0] = sqrtf(rs[0]);
}

__global__ void score_k(const float* __restrict__ h, const float* __restrict__ w,
                        const float* __restrict__ wn, float* __restrict__ score,
                        int Mn, int C, int CP) {
  int i = blockIdx.x * blockDim.x + threadIdx.x;
  if (i >= Mn) return;
  float s = 0.0f;
  for (int c = 0; c < C; ++c) s += h[(size_t)i * CP + c] * w[c];
  score[i] = tanhf(s / wn[0]);
}

__global__ void topk_pool_k(const float* __restrict__ score,
                            const float* __restrict__ hin,
                            float* __restrict__ hout,
                            int* __restrict__ newidx,
                            int n_cur, int k, int CP, int npow2) {
  __shared__ float sval[1024];
  __shared__ int   sidx[1024];
  const int g = blockIdx.x;
  const int t = threadIdx.x;

  if (t < n_cur) newidx[g * n_cur + t] = -1;
  if (t < npow2) {
    if (t < n_cur) { sval[t] = score[g * n_cur + t]; sidx[t] = t; }
    else           { sval[t] = -3.402823e38f;        sidx[t] = t; }
  }
  __syncthreads();

  for (int size = 2; size <= npow2; size <<= 1) {
    for (int stride = size >> 1; stride > 0; stride >>= 1) {
      if (t < npow2) {
        int j = t ^ stride;
        if (j > t) {
          bool descBlk = ((t & size) == 0);
          float vi = sval[t], vj = sval[j];
          int   xi = sidx[t], xj = sidx[j];
          bool iWorse = (vi < vj) || (vi == vj && xi > xj);
          if (iWorse == descBlk) {
            sval[t] = vj; sval[j] = vi;
            sidx[t] = xj; sidx[j] = xi;
          }
        }
      }
      __syncthreads();
    }
  }

  if (t < k) {
    float v = sval[t];
    int old = sidx[t];
    newidx[g * n_cur + old] = g * k + t;
    const float* hr = hin + ((size_t)g * n_cur + old) * CP;
    float* ho = hout + ((size_t)(g * k + t)) * CP;
    for (int c = 0; c < CP; ++c) ho[c] = hr[c] * v;   // padding stays 0
  }
}

// ---------------------------------------------------------------------------
// Readout
// ---------------------------------------------------------------------------
__global__ void global_pool_k(const float* __restrict__ h,
                              float* __restrict__ out3, int n) {
  const int g = blockIdx.x;   // input stride 128, output stride 256
  for (int c = threadIdx.x; c < 256; c += blockDim.x) out3[g * 256 + c] = 0.0f;
  __syncthreads();
  for (int c = threadIdx.x; c < 124; c += blockDim.x) {
    float mx = -3.402823e38f, sm = 0.0f;
    for (int i = 0; i < n; ++i) {
      float v = h[((size_t)g * n + i) * 128 + c];
      mx = fmaxf(mx, v);
      sm += v;
    }
    out3[g * 256 + c]       = mx;
    out3[g * 256 + 124 + c] = sm / (float)n;
  }
}

__global__ void lin3_sigmoid_k(const float* __restrict__ feat,
                               const float* __restrict__ W,
                               const float* __restrict__ b,
                               float* __restrict__ out) {
  int i = blockIdx.x * blockDim.x + threadIdx.x;
  if (i >= BG) return;
  float acc = b[0];
  for (int c = 0; c < 62; ++c) acc += feat[i * 64 + c] * W[c];
  out[i] = 1.0f / (1.0f + expf(-acc));
}

__global__ void copy_feat_k(const float* __restrict__ feat, float* __restrict__ out) {
  int i = blockIdx.x * blockDim.x + threadIdx.x;
  if (i >= BG * 62) return;
  int r = i / 62, c = i - r * 62;
  out[BG + i] = feat[r * 64 + c];
}

// ---------------------------------------------------------------------------
// Host orchestration
// ---------------------------------------------------------------------------
struct PW { __bf16* w; float* b; };

extern "C" void kernel_launch(void* const* d_in, const int* in_sizes, int n_in,
                              void* d_out, int out_size, void* d_ws, size_t ws_size,
                              hipStream_t stream) {
  (void)in_sizes; (void)n_in; (void)out_size; (void)ws_size;

  auto P = [&](int i) { return (const float*)d_in[i]; };
  const float* x  = (const float*)d_in[0];
  const int*   ei = (const int*)d_in[1];
  // Param flattening (pytree sorted keys, recursive), after x/edge_index/batch:
  //  3/4 bn02(beta,g) 5/6 bn1 7/8 bn11 9/10 bn12 11/12 bn2 13/14 bn21 15/16 bn22
  //  conv blocks (Ws,b,key.W,key.b,query.W,query.b,value.W,value.b):
  //  17 conv02, 25 conv11, 33 conv12, 41 conv21, 49 conv22
  //  57/58 lin1  59/60 lin2  61/62 lin3  63 pool1_w 64 pool2_w 65 pool3_w

  char* base = (char*)d_ws;
  size_t off = 0;
  auto alloc = [&](size_t bytes) -> char* {
    char* p = base + off;
    off += (bytes + 255) & ~(size_t)255;
    return p;
  };
  const size_t NMAXP = 3776512;  // max padded rows*channels (59008*64 = 29504*128)
  float* bufA  = (float*)alloc(NMAXP * 4);   // also holds padded x in stage 1
  float* bufB  = (float*)alloc(NMAXP * 4);
  float* fK    = (float*)alloc(NMAXP * 4);
  float* fQ    = (float*)alloc(NMAXP * 4);
  float* fV    = (float*)alloc(NMAXP * 4);
  int*   srcW  = (int*)alloc(EDG * 4);
  int*   dstW  = (int*)alloc(EDG * 4);
  int*   maskW = (int*)alloc(EDG * 4);
  float* scores = (float*)alloc(65536 * 4);
  int*   newidx = (int*)alloc(65536 * 4);
  float* mu    = (float*)alloc(128 * 4);
  float* rstd  = (float*)alloc(128 * 4);
  float* wn    = (float*)alloc(256);
  float* out3  = (float*)alloc(BG * 256 * 4);
  float* z1    = (float*)alloc(BG * 128 * 4);
  float* feat  = (float*)alloc(BG * 64 * 4);

  const int EB = (EDG + 255) / 256;

  // ---- pre-pack weights (bf16 fragment layout) + pad biases -------------
  auto prep = [&](int wi, int bi, int K, int N, int KP, int NP) -> PW {
    const int total = (NP >> 4) * (KP >> 5) * 512;
    __bf16* wp = (__bf16*)alloc((size_t)total * 2);
    float*  bp = (float*)alloc((size_t)NP * 4);
    pack_w_k<<<(total + 255) / 256, 256, 0, stream>>>(P(wi), wp, K, N, KP, NP);
    pad_bias_k<<<1, 256, 0, stream>>>(P(bi), bp, N, NP);
    return PW{wp, bp};
  };
  PW c02W = prep(17, 18, 31, 31, 32, 32);
  PW c02K = prep(19, 20, 31, 31, 32, 32);
  PW c02Q = prep(21, 22, 31, 31, 32, 32);
  PW c02V = prep(23, 24, 31, 31, 32, 32);
  PW c11W = prep(25, 26, 31, 62, 32, 64);
  PW c11K = prep(27, 28, 31, 62, 32, 64);
  PW c11Q = prep(29, 30, 31, 62, 32, 64);
  PW c11V = prep(31, 32, 31, 62, 32, 64);
  PW c12W = prep(33, 34, 62, 62, 64, 64);
  PW c12K = prep(35, 36, 62, 62, 64, 64);
  PW c12Q = prep(37, 38, 62, 62, 64, 64);
  PW c12V = prep(39, 40, 62, 62, 64, 64);
  PW c21W = prep(41, 42, 62, 124, 64, 128);
  PW c21K = prep(43, 44, 62, 124, 64, 128);
  PW c21Q = prep(45, 46, 62, 124, 64, 128);
  PW c21V = prep(47, 48, 62, 124, 64, 128);
  PW c22W = prep(49, 50, 124, 124, 128, 128);
  PW c22K = prep(51, 52, 124, 124, 128, 128);
  PW c22Q = prep(53, 54, 124, 124, 128, 128);
  PW c22V = prep(55, 56, 124, 124, 128, 128);
  PW l1   = prep(57, 58, 248, 124, 256, 128);
  PW l2   = prep(59, 60, 124, 62, 128, 64);

  auto run_gemm = [&](const float* A, PW w, float* Y, int M, int KP, int NP) {
    dim3 grid(NP / 16, M / 64);
    gemm_wmma_bf16<<<grid, 128, 0, stream>>>(A, w.w, w.b, Y, KP, NP);
  };
  auto run_conv = [&](const float* h, int M, int KP, int C, int CP,
                      PW ws_, PW k_, PW q_, PW v_, float* outb) {
    run_gemm(h, k_, fK, M, KP, CP);
    run_gemm(h, q_, fQ, M, KP, CP);
    run_gemm(h, v_, fV, M, KP, CP);
    run_gemm(h, ws_, outb, M, KP, CP);       // h@Ws + b
    edge_aggregate_k<<<EB, 256, 0, stream>>>(fK, fQ, fV, srcW, dstW, maskW,
                                             outb, EDG, C, CP);
  };
  auto run_bn = [&](float* h, int M, int C, int CP,
                    const float* g, const float* beta) {
    bn_stats_k<<<C, 256, 0, stream>>>(h, M, CP, mu, rstd);
    int tot = M * CP;
    bn_apply_relu_k<<<(tot + 255) / 256, 256, 0, stream>>>(h, mu, rstd, g, beta,
                                                           tot, C, CP);
  };
  auto run_pool = [&](const float* hin, float* hout, int n_cur, int k, int C,
                      int CP, int npow2, const float* w) {
    vec_norm_k<<<1, 256, 0, stream>>>(w, C, wn);
    int Mn = BG * n_cur;
    score_k<<<(Mn + 255) / 256, 256, 0, stream>>>(hin, w, wn, scores, Mn, C, CP);
    topk_pool_k<<<BG, 1024, 0, stream>>>(scores, hin, hout, newidx, n_cur, k, CP, npow2);
    edge_remap_k<<<EB, 256, 0, stream>>>(srcW, dstW, maskW, newidx, EDG);
  };

  init_edges_k<<<EB, 256, 0, stream>>>(ei, srcW, dstW, maskW, EDG);
  pad_x_k<<<(BG * NPG * 32 + 255) / 256, 256, 0, stream>>>(x, bufA, BG * NPG * 32);

  // Stage 1
  run_conv(bufA, BG * NPG, 32, 31, 32, c02W, c02K, c02Q, c02V, bufB);
  run_bn(bufB, BG * NPG, 31, 32, P(4), P(3));
  run_pool(bufB, bufA, NPG, N1, 31, 32, 1024, P(63));

  // Stage 2
  run_conv(bufA, BG * N1, 32, 62, 64, c11W, c11K, c11Q, c11V, bufB);
  run_bn(bufB, BG * N1, 62, 64, P(8), P(7));
  run_conv(bufB, BG * N1, 64, 62, 64, c12W, c12K, c12Q, c12V, bufA);
  run_bn(bufA, BG * N1, 62, 64, P(10), P(9));
  run_pool(bufA, bufB, N1, N2, 62, 64, 1024, P(64));

  // Stage 3
  run_conv(bufB, BG * N2, 64, 124, 128, c21W, c21K, c21Q, c21V, bufA);
  run_bn(bufA, BG * N2, 124, 128, P(14), P(13));
  run_conv(bufA, BG * N2, 128, 124, 128, c22W, c22K, c22Q, c22V, bufB);
  run_bn(bufB, BG * N2, 124, 128, P(16), P(15));
  run_pool(bufB, bufA, N2, N3, 124, 128, 512, P(65));

  // Readout
  global_pool_k<<<BG, 128, 0, stream>>>(bufA, out3, N3);
  run_gemm(out3, l1, z1, BG, 256, 128);
  run_bn(z1, BG, 124, 128, P(6), P(5));
  run_gemm(z1, l2, feat, BG, 128, 64);
  run_bn(feat, BG, 62, 64, P(12), P(11));

  float* out = (float*)d_out;
  lin3_sigmoid_k<<<1, 64, 0, stream>>>(feat, P(61), P(62), out);
  copy_feat_k<<<(BG * 62 + 255) / 256, 256, 0, stream>>>(feat, out);
}
